// Quantizer1d_16870631539146
// MI455X (gfx1250) — matrix operand order repaired and checked by reading, not compiled, not run on hardware
//
#include <hip/hip_runtime.h>
#include <hip/hip_bf16.h>
#include <math.h>

// ---- problem constants (from reference) ----
#define G_   8
#define D_   128
#define CB_  1024
#define B_   16
#define C_   32
#define T_   16384
#define S_   128
#define N_BH 512            // K*S rows per (b,h)
#define EPS_ 1e-10f

// output layout (flat, concatenated in return order, all as float)
#define XOUT_OFF   0
#define XOUT_SZ    (B_*C_*T_)            // 8388608
#define IDX_OFF    (XOUT_SZ)             // 8388608
#define IDX_SZ     (B_*C_*S_)            // 65536
#define PERP_OFF   (IDX_OFF + IDX_SZ)    // 8454144
#define MASKI_OFF  (PERP_OFF + G_)       // 8454152

// main kernel geometry: 32 n-rows per block (halves kk L2 traffic vs 16)
#define ROWS_  32
#define LDSA_STRIDE 132                  // 128 + 4 dword pad per row (bank rotate)

typedef __attribute__((ext_vector_type(16))) __bf16 v16bf;
typedef __attribute__((ext_vector_type(8)))  float  v8f;
typedef __attribute__((ext_vector_type(4)))  unsigned int v4u;
typedef __attribute__((ext_vector_type(8)))  int v8i;
typedef __attribute__((ext_vector_type(4)))  int v4i;

union FragBF {
  v16bf  v;
  __bf16 e[16];
  float4 f4[2];
};

__device__ inline v8f wmma_bf16(const FragBF& a, const FragBF& b, v8f c) {
  // emits v_wmma_f32_16x16x32_bf16
  return __builtin_amdgcn_wmma_f32_16x16x32_bf16(
      false, a.v, false, b.v, (short)0, c, false, false);
}

__device__ inline void cvt_hi_lo8(const float* f, __bf16* hi, __bf16* lo) {
#pragma unroll
  for (int i = 0; i < 8; ++i) {
    __bf16 h = (__bf16)f[i];
    hi[i] = h;
    lo[i] = (__bf16)(f[i] - (float)h);
  }
}

// ---------------------------------------------------------------------------
// Tensor Data Mover: 2-D tile load to LDS (f32 elements).
//   tensor/tile dims in elements; optional LDS padding: pad_amount_code+1
//   dwords inserted every 2^(pad_interval_code+1) dwords.
// ---------------------------------------------------------------------------
#if __has_builtin(__builtin_amdgcn_tensor_load_to_lds)
#define HAVE_TDM 1
__device__ inline void tdm_load_2d_f32(unsigned int lds_off, const void* gaddr,
                                       int d0, int d1, int stride0,
                                       int pad_en, int pad_int_code, int pad_amt_code) {
  unsigned long long ga = (unsigned long long)gaddr;
  v4u g0 = { 1u,                                   // count=1, user descriptor
             lds_off,                              // lds_addr
             (unsigned)(ga & 0xffffffffu),         // global_addr[31:0]
             (unsigned)((ga >> 32) & 0x01ffffffu)  // global_addr[56:32]
               | 0x80000000u };                    // type=2 ("image")
  int w0 = 0x00020000 | (pad_en << 20) | (pad_int_code << 22) | (pad_amt_code << 25);
  v8i g1 = { w0,                                          // data_size=4B, pad cfg
             (int)((unsigned)d0 << 16),                   // tensor_dim0[15:0]
             (int)(((unsigned)d0 >> 16) | ((unsigned)d1 << 16)),  // dim0 hi | dim1 lo
             (int)(((unsigned)d1 >> 16) | ((unsigned)d0 << 16)),  // dim1 hi | tile_dim0
             (int)((unsigned)d1 & 0xffffu),               // tile_dim1 (tile==tensor)
             stride0,                                     // tensor_dim0_stride[31:0]
             0, 0 };
  v4i gz = {0, 0, 0, 0};
#if __clang_major__ >= 23
  v8i gz8 = {0, 0, 0, 0, 0, 0, 0, 0};
  __builtin_amdgcn_tensor_load_to_lds(g0, g1, gz, gz, gz8, 0);
#else
  __builtin_amdgcn_tensor_load_to_lds(g0, g1, gz, gz, 0);
#endif
}
#else
#define HAVE_TDM 0
#endif

// ---------------------------------------------------------------------------
// Kernel 0: zero histogram workspace (ws is poisoned, must re-init each call)
// ---------------------------------------------------------------------------
__global__ void zero_hist_kernel(int* __restrict__ hist) {
  int i = blockIdx.x * blockDim.x + threadIdx.x;
  if (i < G_ * CB_) hist[i] = 0;
}

// ---------------------------------------------------------------------------
// Kernel 1: per-head GEMM  out = codebooks[h] (CBxD) @ W[h] (DxD)
// mode 0: store bf16 hi/lo split (kk);  mode 1: store f32 (v)
// ---------------------------------------------------------------------------
__global__ __launch_bounds__(256)
void prep_gemm_kernel(const float* __restrict__ cb,   // [G,CB,D]
                      const float* __restrict__ W,    // [G,D,D]
                      __bf16* __restrict__ out_hi,
                      __bf16* __restrict__ out_lo,
                      float*  __restrict__ out_f32,
                      int mode) {
  __shared__ __bf16 Wt_hi[D_ * D_];   // W transposed: Wt[n][d] = W[d][n]
  __shared__ __bf16 Wt_lo[D_ * D_];   // 64 KB

  const int h     = blockIdx.x >> 6;
  const int mtile = blockIdx.x & 63;
  const int tid   = threadIdx.x;

  const float* Wh = W + (size_t)h * D_ * D_;
  for (int e = tid; e < D_ * D_; e += 256) {
    int d = e >> 7, n = e & 127;
    float f = Wh[e];
    __bf16 hb = (__bf16)f;
    Wt_hi[n * D_ + d] = hb;
    Wt_lo[n * D_ + d] = (__bf16)(f - (float)hb);
  }
  __syncthreads();

  const int wave = tid >> 5, lane = tid & 31;
  const int m = lane & 15, hiL = lane >> 4;
  const int jb = wave * 16;

  const float* arow = cb + ((size_t)(h * CB_ + mtile * 16 + m)) * D_;
  FragBF a_hi[4], a_lo[4];
#pragma unroll
  for (int kc = 0; kc < 4; ++kc) {
    float tmp[16];
    const float* p0 = arow + kc * 32 + hiL * 16;
    const float* p1 = p0 + 8;
#pragma unroll
    for (int i = 0; i < 8; ++i) { tmp[i] = p0[i]; tmp[8 + i] = p1[i]; }
    cvt_hi_lo8(tmp,     &a_hi[kc].e[0], &a_lo[kc].e[0]);
    cvt_hi_lo8(tmp + 8, &a_hi[kc].e[8], &a_lo[kc].e[8]);
  }

  v8f acc = {0.f, 0.f, 0.f, 0.f, 0.f, 0.f, 0.f, 0.f};
#pragma unroll
  for (int kc = 0; kc < 4; ++kc) {
    FragBF b_hi, b_lo;
    const __bf16* bh = &Wt_hi[(jb + m) * D_ + kc * 32 + hiL * 16];
    const __bf16* bl = &Wt_lo[(jb + m) * D_ + kc * 32 + hiL * 16];
#pragma unroll
    for (int i = 0; i < 16; ++i) { b_hi.e[i] = bh[i]; b_lo.e[i] = bl[i]; }
    acc = wmma_bf16(a_hi[kc], b_hi, acc);
    acc = wmma_bf16(a_hi[kc], b_lo, acc);
    acc = wmma_bf16(a_lo[kc], b_hi, acc);
  }

  const int col = jb + m;
#pragma unroll
  for (int r = 0; r < 8; ++r) {
    const int row = mtile * 16 + r + hiL * 8;
    const size_t off = ((size_t)(h * CB_ + row)) * D_ + col;
    const float f = acc[r];
    if (mode == 0) {
      __bf16 hb = (__bf16)f;
      out_hi[off] = hb;
      out_lo[off] = (__bf16)(f - (float)hb);
    } else {
      out_f32[off] = f;
    }
  }
}

// ---------------------------------------------------------------------------
// Kernel 2: main fused quantizer. One block per (b, h, 32-row n-tile).
//   A slab (32x128 f32, contiguous) pulled into LDS by the TDM with row
//   padding; logits via bf16-split WMMA with B reused across 2 row tiles;
//   fused gumbel+argmax on the C/D fragment; hist; masked v-gather writes.
// ---------------------------------------------------------------------------
__global__ __launch_bounds__(256)
void quantize_main_kernel(const float* __restrict__ x,
                          const __bf16* __restrict__ kk_hi,
                          const __bf16* __restrict__ kk_lo,
                          const float* __restrict__ vmat,
                          const float* __restrict__ fixed_tokens,
                          const float* __restrict__ gumbel,
                          const unsigned char* __restrict__ mask,
                          float* __restrict__ out,
                          int* __restrict__ hist) {
  const int blk   = blockIdx.x;
  const int b     = blk >> 7;          // 16
  const int h     = (blk >> 4) & 7;    // 8
  const int ntile = blk & 15;          // 16 tiles of 32 rows over n=512
  const int kq    = ntile >> 2;        // n = kq*S + s
  const int sbase = (ntile & 3) * ROWS_;
  const int n0    = ntile * ROWS_;

  const int tid  = threadIdx.x;
  const int wave = tid >> 5, lane = tid & 31;
  const int m = lane & 15, hiL = lane >> 4;

  __shared__ float ldsA[ROWS_ * LDSA_STRIDE];   // padded 32x128 A slab
  __shared__ float red_val[8 * ROWS_];
  __shared__ int   red_idx[8 * ROWS_];
  __shared__ int   fidx[ROWS_];

  // ---- stage A slab (contiguous 32x128 f32 block of x) into LDS ----
  const float* xblk = x + ((size_t)(b * C_ + h * 4 + kq)) * T_ + (size_t)sbase * D_;
#if HAVE_TDM
  if (wave == 0) {
    // rows of 128 dwords; insert 4 dwords after every 128 -> LDS stride 132
    tdm_load_2d_f32((unsigned)(uintptr_t)&ldsA[0], xblk,
                    /*d0=*/D_, /*d1=*/ROWS_, /*stride0=*/D_,
                    /*pad_en=*/1, /*pad_int(2^7=128)=*/6, /*pad_amt(4dw)=*/3);
    __builtin_amdgcn_s_wait_tensorcnt(0);
  }
  __syncthreads();
#else
  for (int e = tid; e < ROWS_ * D_; e += 256) {
    int r = e >> 7, c = e & 127;
    ldsA[r * LDSA_STRIDE + c] = xblk[(size_t)r * D_ + c];
  }
  __syncthreads();
#endif

  // ---- build A fragments (both row tiles) from LDS, bf16 hi/lo split ----
  const float scale = 0.08838834764831845f;  // D^-0.5
  FragBF a_hi[2][4], a_lo[2][4];
#pragma unroll
  for (int rt = 0; rt < 2; ++rt) {
    const float* arow = &ldsA[(rt * 16 + m) * LDSA_STRIDE];
#pragma unroll
    for (int kc = 0; kc < 4; ++kc) {
      float tmp[16];
      const float* p0 = arow + kc * 32 + hiL * 16;
      const float* p1 = p0 + 8;
#pragma unroll
      for (int i = 0; i < 8; ++i) { tmp[i] = p0[i] * scale; tmp[8 + i] = p1[i] * scale; }
      cvt_hi_lo8(tmp,     &a_hi[rt][kc].e[0], &a_lo[rt][kc].e[0]);
      cvt_hi_lo8(tmp + 8, &a_hi[rt][kc].e[8], &a_lo[rt][kc].e[8]);
    }
  }

  const float* grow = gumbel + (((size_t)(b * G_ + h)) * N_BH + n0) * CB_;

  float bv[2][8]; int bi[2][8];
#pragma unroll
  for (int rt = 0; rt < 2; ++rt)
#pragma unroll
    for (int r = 0; r < 8; ++r) { bv[rt][r] = -__builtin_inff(); bi[rt][r] = 0; }

  // each wave sweeps 8 of the 64 CB column tiles; B loaded once per k-chunk
  for (int t = 0; t < 8; ++t) {
    const int jb = (wave * 8 + t) * 16;
    __builtin_prefetch(grow + (size_t)hiL * 8 * CB_ + jb + 16 + m, 0, 1);

    v8f acc[2];
    acc[0] = (v8f){0.f, 0.f, 0.f, 0.f, 0.f, 0.f, 0.f, 0.f};
    acc[1] = (v8f){0.f, 0.f, 0.f, 0.f, 0.f, 0.f, 0.f, 0.f};
    const size_t brow = ((size_t)(h * CB_ + jb + m)) * D_;
#pragma unroll
    for (int kc = 0; kc < 4; ++kc) {
      FragBF b_hi, b_lo;
      const float4* ph = (const float4*)(kk_hi + brow + kc * 32 + hiL * 16);
      const float4* pl = (const float4*)(kk_lo + brow + kc * 32 + hiL * 16);
      b_hi.f4[0] = ph[0]; b_hi.f4[1] = ph[1];
      b_lo.f4[0] = pl[0]; b_lo.f4[1] = pl[1];
#pragma unroll
      for (int rt = 0; rt < 2; ++rt) {
        acc[rt] = wmma_bf16(a_hi[rt][kc], b_hi, acc[rt]);
        acc[rt] = wmma_bf16(a_hi[rt][kc], b_lo, acc[rt]);
        acc[rt] = wmma_bf16(a_lo[rt][kc], b_hi, acc[rt]);
      }
    }
    // fused gumbel add + running argmax directly on the C/D fragments
#pragma unroll
    for (int rt = 0; rt < 2; ++rt) {
#pragma unroll
      for (int r = 0; r < 8; ++r) {
        const int M = rt * 16 + r + hiL * 8;
        const float val = acc[rt][r] + grow[(size_t)M * CB_ + jb + m];
        const int   idx = jb + m;
        if (val > bv[rt][r]) { bv[rt][r] = val; bi[rt][r] = idx; }
      }
    }
  }

  // reduce across the 16 lanes of each half-wave (wave32 shuffles)
#pragma unroll
  for (int rt = 0; rt < 2; ++rt) {
#pragma unroll
    for (int r = 0; r < 8; ++r) {
      float v = bv[rt][r]; int i = bi[rt][r];
#pragma unroll
      for (int off = 8; off; off >>= 1) {
        float ov = __shfl_xor(v, off, 32);
        int   oi = __shfl_xor(i, off, 32);
        if (ov > v || (ov == v && oi < i)) { v = ov; i = oi; }
      }
      bv[rt][r] = v; bi[rt][r] = i;
    }
  }
  if (m == 0) {
#pragma unroll
    for (int rt = 0; rt < 2; ++rt)
#pragma unroll
      for (int r = 0; r < 8; ++r) {
        red_val[wave * ROWS_ + rt * 16 + hiL * 8 + r] = bv[rt][r];
        red_idx[wave * ROWS_ + rt * 16 + hiL * 8 + r] = bi[rt][r];
      }
  }
  __syncthreads();

  // final cross-wave argmax + scalar outputs
  if (tid < ROWS_) {
    const int M = tid;
    float v = red_val[M]; int i = red_idx[M];
#pragma unroll
    for (int w = 1; w < 8; ++w) {
      float ov = red_val[w * ROWS_ + M];
      int   oi = red_idx[w * ROWS_ + M];
      if (ov > v || (ov == v && oi < i)) { v = ov; i = oi; }
    }
    fidx[M] = i;
    atomicAdd(&hist[h * CB_ + i], 1);

    const int s = sbase + M;
    const int cp = h * 4 + kq;
    const size_t io = (size_t)(b * C_ + cp) * S_ + s;
    out[IDX_OFF + io]   = (float)i;
    out[MASKI_OFF + io] = mask[(size_t)(b * S_ + s) * C_ + cp] ? 1.0f : 0.0f;
  }
  __syncthreads();

  // masked gather of v rows -> x_out (32 rows x 128 f32, contiguous segments)
  {
    const int M = tid >> 3;            // row 0..31
    const int p = tid & 7;             // 16 floats per thread
    const int s = sbase + M;
    const int cp = h * 4 + kq;
    const bool mk = mask[(size_t)(b * S_ + s) * C_ + cp] != 0;
    const float* src = mk ? (vmat + ((size_t)(h * CB_ + fidx[M])) * D_ + p * 16)
                          : (fixed_tokens + (size_t)cp * D_ + p * 16);
    float* dst = out + ((size_t)(b * C_ + cp)) * T_ + (size_t)s * D_ + p * 16;
#pragma unroll
    for (int q = 0; q < 4; ++q)
      ((float4*)dst)[q] = ((const float4*)src)[q];
  }
}

// ---------------------------------------------------------------------------
// Kernel 3: histogram -> perplexity per head
// ---------------------------------------------------------------------------
__global__ __launch_bounds__(256)
void perp_kernel(const int* __restrict__ hist, float* __restrict__ out_perp) {
  const int h = blockIdx.x;
  const int tid = threadIdx.x;
  float s = 0.f;
  for (int j = tid; j < CB_; j += 256) {
    const float p = (float)hist[h * CB_ + j] * (1.0f / (float)(B_ * N_BH));
    s += p * logf(p + EPS_);
  }
  __shared__ float red[8];
#pragma unroll
  for (int off = 16; off; off >>= 1) s += __shfl_xor(s, off, 32);
  if ((tid & 31) == 0) red[tid >> 5] = s;
  __syncthreads();
  if (tid == 0) {
    float tot = 0.f;
#pragma unroll
    for (int w = 0; w < 8; ++w) tot += red[w];
    out_perp[h] = expf(-tot);
  }
}

// ---------------------------------------------------------------------------
extern "C" void kernel_launch(void* const* d_in, const int* in_sizes, int n_in,
                              void* d_out, int out_size, void* d_ws, size_t ws_size,
                              hipStream_t stream) {
  const float* x            = (const float*)d_in[0];
  const float* codebooks    = (const float*)d_in[1];
  const float* Wk           = (const float*)d_in[2];
  const float* Wv           = (const float*)d_in[3];
  const float* fixed_tokens = (const float*)d_in[4];
  const float* gumbel       = (const float*)d_in[5];
  const unsigned char* mask = (const unsigned char*)d_in[6];
  float* out = (float*)d_out;

  // workspace: kk_hi (2MB) | kk_lo (2MB) | v f32 (4MB) | hist (32KB)
  char* ws = (char*)d_ws;
  const size_t NE = (size_t)G_ * CB_ * D_;     // 1,048,576 elements
  __bf16* kk_hi = (__bf16*)(ws);
  __bf16* kk_lo = (__bf16*)(ws + NE * 2);
  float*  vmat  = (float*) (ws + NE * 4);
  int*    hist  = (int*)   (ws + NE * 8);

  zero_hist_kernel<<<(G_ * CB_ + 255) / 256, 256, 0, stream>>>(hist);
  prep_gemm_kernel<<<G_ * (CB_ / 16), 256, 0, stream>>>(
      codebooks, Wk, kk_hi, kk_lo, nullptr, 0);
  prep_gemm_kernel<<<G_ * (CB_ / 16), 256, 0, stream>>>(
      codebooks, Wv, nullptr, nullptr, vmat, 1);
  quantize_main_kernel<<<B_ * G_ * 16, 256, 0, stream>>>(
      x, kk_hi, kk_lo, vmat, fixed_tokens, gumbel, mask, out, hist);
  perp_kernel<<<G_, 256, 0, stream>>>(hist, out + PERP_OFF);
}
